// ScaledDotProduction_23759759081831
// MI455X (gfx1250) — compile-verified, matrix operand order checked by reading, not run on hardware
//
#include <hip/hip_runtime.h>

// ---- CDNA5 WMMA types ----
typedef __attribute__((ext_vector_type(16))) __bf16 v16bf;
typedef __attribute__((ext_vector_type(8)))  float  v8f;
typedef __attribute__((ext_vector_type(8)))  __bf16 bf16x8;
typedef __attribute__((ext_vector_type(4)))  __bf16 bf16x4;

#define B_SZ 32
#define N_SZ 2048
#define D_SZ 128
#define KT   64            // keys per tile (bigger tile => less softmax overhead per WMMA)
#define QW   16            // query rows per wave
#define WAVES 4
#define QB   (QW * WAVES)  // 64 query rows per block
#define NKT  (N_SZ / KT)   // 32 key tiles
#define STILES (KT / 16)   // 4 score tiles per wave
#define KCH  (KT / 32)     // 2 PV k-chunks
#define DT   (D_SZ / 16)   // 8 output d-tiles

__device__ __forceinline__ v16bf pack16(bf16x8 lo, bf16x8 hi) {
    v16bf r;
#pragma unroll
    for (int i = 0; i < 8; ++i) { r[i] = lo[i]; r[i + 8] = hi[i]; }
    return r;
}

// CDNA5 async DMA: global -> LDS, 128b per lane, tracked by ASYNCcnt (no VGPR data).
__device__ __forceinline__ void async_b128(unsigned lds_off, unsigned long long gaddr) {
    asm volatile("global_load_async_to_lds_b128 %0, %1, off"
                 :: "v"(lds_off), "v"(gaddr) : "memory");
}
__device__ __forceinline__ void wait_async0() {
    asm volatile("s_wait_asynccnt 0x0" ::: "memory");
}

// ---------------------------------------------------------------------------
// Pre-pass: fp32 -> bf16 once over HBM (compute-bound problem).
// Q is pre-scaled by 1/sqrt(D). V is stored tile-transposed:
//   Vh[((b*NKT + kt)*D + d)*KT + key]  => every (b,kt) tile is a contiguous
//   16KB [D][KT] block, DMA-able straight into LDS with no shuffling.
// K stays row-major: each (b,kt) tile is a contiguous 16KB [KT][D] block.
// ---------------------------------------------------------------------------
__global__ __launch_bounds__(256)
void convert_kernel(const float* __restrict__ Q, const float* __restrict__ K,
                    const float* __restrict__ V,
                    __bf16* __restrict__ Qh, __bf16* __restrict__ Kh,
                    __bf16* __restrict__ Vh)
{
    const float scale = 0.08838834764831845f;  // 1/sqrt(128)
    size_t g = (size_t)blockIdx.x * 256 + threadIdx.x;
    size_t base = g * 8;

    {   // Q (scaled) and K: same linear layout
        float4 a0 = ((const float4*)(Q + base))[0];
        float4 a1 = ((const float4*)(Q + base))[1];
        bf16x8 rq;
        rq[0] = (__bf16)(a0.x * scale); rq[1] = (__bf16)(a0.y * scale);
        rq[2] = (__bf16)(a0.z * scale); rq[3] = (__bf16)(a0.w * scale);
        rq[4] = (__bf16)(a1.x * scale); rq[5] = (__bf16)(a1.y * scale);
        rq[6] = (__bf16)(a1.z * scale); rq[7] = (__bf16)(a1.w * scale);
        *(bf16x8*)(Qh + base) = rq;

        float4 b0 = ((const float4*)(K + base))[0];
        float4 b1 = ((const float4*)(K + base))[1];
        bf16x8 rk;
        rk[0] = (__bf16)b0.x; rk[1] = (__bf16)b0.y;
        rk[2] = (__bf16)b0.z; rk[3] = (__bf16)b0.w;
        rk[4] = (__bf16)b1.x; rk[5] = (__bf16)b1.y;
        rk[6] = (__bf16)b1.z; rk[7] = (__bf16)b1.w;
        *(bf16x8*)(Kh + base) = rk;
    }
    {   // V: tile-transposed
        size_t bb = base / ((size_t)N_SZ * D_SZ);
        size_t r  = base % ((size_t)N_SZ * D_SZ);
        int n = (int)(r / D_SZ);
        int d = (int)(r % D_SZ);
        int ktl = n / KT, key = n % KT;
        float4 v0 = ((const float4*)(V + base))[0];
        float4 v1 = ((const float4*)(V + base))[1];
        __bf16* dst = Vh + (((size_t)bb * NKT + ktl) * D_SZ + d) * KT + key;
        dst[0 * KT] = (__bf16)v0.x; dst[1 * KT] = (__bf16)v0.y;
        dst[2 * KT] = (__bf16)v0.z; dst[3 * KT] = (__bf16)v0.w;
        dst[4 * KT] = (__bf16)v1.x; dst[5 * KT] = (__bf16)v1.y;
        dst[6 * KT] = (__bf16)v1.z; dst[7 * KT] = (__bf16)v1.w;
    }
}

// ---------------------------------------------------------------------------
// Flash attention. FROM_WS=true: bf16 operands from workspace, K/V tiles
// double-buffered in LDS via async DMA overlapped with compute.
// FROM_WS=false: self-contained fallback with fused fp32->bf16 staging.
// ---------------------------------------------------------------------------
template<bool FROM_WS>
__global__ __launch_bounds__(128)
void attn_kernel(const float* __restrict__ Qf32, const float* __restrict__ Kf32,
                 const float* __restrict__ Vf32,
                 const __bf16* __restrict__ Qh, const __bf16* __restrict__ Kh,
                 const __bf16* __restrict__ Vh, float* __restrict__ O)
{
    __shared__ __bf16 Ks[2][KT][D_SZ];     // 32 KB
    __shared__ __bf16 Vt[2][D_SZ][KT];     // 32 KB
    __shared__ __bf16 Pb[WAVES][QW][KT];   //  8 KB

    const int tid  = threadIdx.x;
    const int w    = tid >> 5;
    const int lane = tid & 31;
    const int hf   = lane >> 4;
    const int nc   = lane & 15;

    const int blk  = blockIdx.x;
    const int b    = blk / (N_SZ / QB);
    const int qblk = blk % (N_SZ / QB);
    const int q0   = qblk * QB + w * QW;

    const float scale = 0.08838834764831845f;

    // ---- Q fragments (A layout), resident for whole kernel ----
    v16bf qf[4];
    if constexpr (FROM_WS) {
        const __bf16* qrow = Qh + (size_t)(b * N_SZ + q0 + nc) * D_SZ;
#pragma unroll
        for (int ks = 0; ks < 4; ++ks) {
            bf16x8 lo = *(const bf16x8*)(qrow + ks * 32 + hf * 8);
            bf16x8 hi = *(const bf16x8*)(qrow + ks * 32 + 16 + hf * 8);
            qf[ks] = pack16(lo, hi);
        }
    } else {
        const float* qrow = Qf32 + (size_t)(b * N_SZ + q0 + nc) * D_SZ;
#pragma unroll
        for (int ks = 0; ks < 4; ++ks) {
            const float* g0 = qrow + ks * 32 + hf * 8;
            const float* g1 = qrow + ks * 32 + 16 + hf * 8;
            float4 A0 = ((const float4*)g0)[0];
            float4 A1 = ((const float4*)g0)[1];
            float4 B0 = ((const float4*)g1)[0];
            float4 B1 = ((const float4*)g1)[1];
            v16bf f;
            f[0]  = (__bf16)(A0.x * scale); f[1]  = (__bf16)(A0.y * scale);
            f[2]  = (__bf16)(A0.z * scale); f[3]  = (__bf16)(A0.w * scale);
            f[4]  = (__bf16)(A1.x * scale); f[5]  = (__bf16)(A1.y * scale);
            f[6]  = (__bf16)(A1.z * scale); f[7]  = (__bf16)(A1.w * scale);
            f[8]  = (__bf16)(B0.x * scale); f[9]  = (__bf16)(B0.y * scale);
            f[10] = (__bf16)(B0.z * scale); f[11] = (__bf16)(B0.w * scale);
            f[12] = (__bf16)(B1.x * scale); f[13] = (__bf16)(B1.y * scale);
            f[14] = (__bf16)(B1.z * scale); f[15] = (__bf16)(B1.w * scale);
            qf[ks] = f;
        }
    }

    // issue one tile's async DMA: 16KB K + 16KB Vt, 8+8 b128 chunks per thread
    auto issue_tile = [&](int kt, int bufn) {
        unsigned long long kg = (unsigned long long)(uintptr_t)
            (Kh + (size_t)(b * N_SZ + kt * KT) * D_SZ);
        unsigned long long vg = (unsigned long long)(uintptr_t)
            (Vh + ((size_t)(b * NKT + kt)) * D_SZ * KT);
        unsigned kbase = (unsigned)(uintptr_t)(&Ks[bufn][0][0]);
        unsigned vbase = (unsigned)(uintptr_t)(&Vt[bufn][0][0]);
        unsigned off = (unsigned)tid * 16u;
#pragma unroll
        for (int i = 0; i < 8; ++i) {
            async_b128(kbase + off, kg + off);
            async_b128(vbase + off, vg + off);
            off += 128u * 16u;
        }
    };

    // ---- running state ----
    v8f acc[DT];
#pragma unroll
    for (int t = 0; t < DT; ++t)
#pragma unroll
        for (int i = 0; i < 8; ++i) acc[t][i] = 0.0f;
    float m_i[8], l_i[8];
#pragma unroll
    for (int i = 0; i < 8; ++i) { m_i[i] = -1.0e30f; l_i[i] = 0.0f; }

    if constexpr (FROM_WS) issue_tile(0, 0);

    for (int kt = 0; kt < NKT; ++kt) {
        const int bufn = kt & 1;

        if constexpr (FROM_WS) {
            wait_async0();        // this wave's current-tile DMA landed
            __syncthreads();      // everyone's landed; prior buf^1 reads done
            if (kt + 1 < NKT) issue_tile(kt + 1, bufn ^ 1);  // overlap next DMA
        } else {
            __syncthreads();      // previous iteration's LDS reads done
            const int key = tid >> 1;
            const int db  = (tid & 1) * 64;
            const float* kg = Kf32 + (size_t)(b * N_SZ + kt * KT + key) * D_SZ + db;
            const float* vg = Vf32 + (size_t)(b * N_SZ + kt * KT + key) * D_SZ + db;
#pragma unroll
            for (int i = 0; i < 16; ++i) {
                float4 k4 = ((const float4*)kg)[i];
                bf16x4 h;
                h[0] = (__bf16)k4.x; h[1] = (__bf16)k4.y;
                h[2] = (__bf16)k4.z; h[3] = (__bf16)k4.w;
                *(bf16x4*)&Ks[bufn][key][db + i * 4] = h;
                float4 v4 = ((const float4*)vg)[i];
                Vt[bufn][db + i * 4 + 0][key] = (__bf16)v4.x;
                Vt[bufn][db + i * 4 + 1][key] = (__bf16)v4.y;
                Vt[bufn][db + i * 4 + 2][key] = (__bf16)v4.z;
                Vt[bufn][db + i * 4 + 3][key] = (__bf16)v4.w;
            }
            if (kt + 1 < NKT) {
                __builtin_prefetch(kg + KT * D_SZ, 0, 0);
                __builtin_prefetch(vg + KT * D_SZ, 0, 0);
            }
            __syncthreads();
        }

        // ---- S = (Q*scale) @ K^T : 4 x 16x16 f32 tiles over 64 keys ----
        v8f s[STILES];
#pragma unroll
        for (int t = 0; t < STILES; ++t)
#pragma unroll
            for (int i = 0; i < 8; ++i) s[t][i] = 0.0f;

#pragma unroll
        for (int ks = 0; ks < 4; ++ks) {
#pragma unroll
            for (int t = 0; t < STILES; ++t) {
                bf16x8 lo = *(const bf16x8*)&Ks[bufn][t * 16 + nc][ks * 32 + hf * 8];
                bf16x8 hi = *(const bf16x8*)&Ks[bufn][t * 16 + nc][ks * 32 + 16 + hf * 8];
                s[t] = __builtin_amdgcn_wmma_f32_16x16x32_bf16(
                    false, qf[ks], false, pack16(lo, hi), (short)0, s[t], false, false);
            }
        }

        // ---- online softmax over 64 keys (C layout: row = v+8*hf, col = nc) ----
#pragma unroll
        for (int v = 0; v < 8; ++v) {
            float x = fmaxf(fmaxf(s[0][v], s[1][v]), fmaxf(s[2][v], s[3][v]));
#pragma unroll
            for (int off = 8; off >= 1; off >>= 1)
                x = fmaxf(x, __shfl_xor(x, off, 32));
            float mn = fmaxf(m_i[v], x);
            float al = __expf(m_i[v] - mn);
            float p0 = __expf(s[0][v] - mn);
            float p1 = __expf(s[1][v] - mn);
            float p2 = __expf(s[2][v] - mn);
            float p3 = __expf(s[3][v] - mn);
            float rs = (p0 + p1) + (p2 + p3);
#pragma unroll
            for (int off = 8; off >= 1; off >>= 1)
                rs += __shfl_xor(rs, off, 32);
            l_i[v] = l_i[v] * al + rs;
            m_i[v] = mn;
#pragma unroll
            for (int t = 0; t < DT; ++t) acc[t][v] *= al;
            Pb[w][v + 8 * hf][0 * 16 + nc] = (__bf16)p0;
            Pb[w][v + 8 * hf][1 * 16 + nc] = (__bf16)p1;
            Pb[w][v + 8 * hf][2 * 16 + nc] = (__bf16)p2;
            Pb[w][v + 8 * hf][3 * 16 + nc] = (__bf16)p3;
        }
        __syncthreads();  // P scratch visible across lanes

        // ---- P fragments (A layout), one per 32-key chunk ----
        v16bf pf[KCH];
#pragma unroll
        for (int c = 0; c < KCH; ++c) {
            bf16x8 lo = *(const bf16x8*)&Pb[w][nc][c * 32 + hf * 8];
            bf16x8 hi = *(const bf16x8*)&Pb[w][nc][c * 32 + 16 + hf * 8];
            pf[c] = pack16(lo, hi);
        }

        // ---- O += P @ V : 8 d-tiles x 2 k-chunks ----
#pragma unroll
        for (int t = 0; t < DT; ++t) {
#pragma unroll
            for (int c = 0; c < KCH; ++c) {
                bf16x8 lo = *(const bf16x8*)&Vt[bufn][t * 16 + nc][c * 32 + hf * 8];
                bf16x8 hi = *(const bf16x8*)&Vt[bufn][t * 16 + nc][c * 32 + 16 + hf * 8];
                acc[t] = __builtin_amdgcn_wmma_f32_16x16x32_bf16(
                    false, pf[c], false, pack16(lo, hi), (short)0, acc[t], false, false);
            }
        }
    }

    // ---- epilogue: normalize, store fp32 ----
#pragma unroll
    for (int v = 0; v < 8; ++v) {
        float inv = 1.0f / l_i[v];
        float* orow = O + (size_t)(b * N_SZ + q0 + v + 8 * hf) * D_SZ;
#pragma unroll
        for (int t = 0; t < DT; ++t)
            orow[t * 16 + nc] = acc[t][v] * inv;
    }
}

extern "C" void kernel_launch(void* const* d_in, const int* in_sizes, int n_in,
                              void* d_out, int out_size, void* d_ws, size_t ws_size,
                              hipStream_t stream) {
    const float* q = (const float*)d_in[0];
    const float* k = (const float*)d_in[1];
    const float* v = (const float*)d_in[2];
    float* out = (float*)d_out;

    const size_t elems = (size_t)B_SZ * N_SZ * D_SZ;     // 8.39M per tensor
    const size_t need  = elems * 2 * 3;                  // bf16 Q,K,V ~50MB
    dim3 agrid(B_SZ * (N_SZ / QB));                      // 1024 blocks
    dim3 ablock(128);                                    // 4 waves (wave32)

    if (ws_size >= need) {
        __bf16* Qh = (__bf16*)d_ws;
        __bf16* Kh = Qh + elems;
        __bf16* Vh = Kh + elems;
        dim3 cgrid((unsigned)(elems / 8 / 256));         // 4096 blocks
        convert_kernel<<<cgrid, 256, 0, stream>>>(q, k, v, Qh, Kh, Vh);
        attn_kernel<true><<<agrid, ablock, 0, stream>>>(q, k, v, Qh, Kh, Vh, out);
    } else {
        attn_kernel<false><<<agrid, ablock, 0, stream>>>(q, k, v,
                                                         nullptr, nullptr, nullptr, out);
    }
}
